// MLPGConv_Z2_38319698215298
// MI455X (gfx1250) — matrix operand order repaired and checked by reading, not compiled, not run on hardware
//
#include <hip/hip_runtime.h>
#include <hip/hip_bf16.h>
#include <stdint.h>

// ---- problem constants ----
#define B_SZ   32
#define CIN    32
#define COUT   64
#define P_SZ   1024   // IMG*IMG
#define Q_SZ   784    // OUT*OUT
#define K2_SZ  25
#define M_SZ   64
#define NBLK   64
#define KIL    2048   // CIN*NBLK
#define NKO    256    // 4*COUT
#define MBQ    25088  // B*Q

#define LDSPAD 20     // dword row stride (32 bf16 = 16 dwords used + 4 pad) -> conflict-free

typedef __attribute__((ext_vector_type(16))) __bf16        v16bf;
typedef __attribute__((ext_vector_type(8)))  float         v8f;
typedef __attribute__((ext_vector_type(8)))  unsigned int  v8u;
typedef __attribute__((ext_vector_type(4)))  int           v4i_t;

#define AS1 __attribute__((address_space(1)))
#define AS3 __attribute__((address_space(3)))

#if __has_builtin(__builtin_amdgcn_global_load_async_to_lds_b128) && \
    __has_builtin(__builtin_amdgcn_s_wait_asynccnt)
#define HAVE_ASYNC 1
#define WAIT_ASYNC(n) __builtin_amdgcn_s_wait_asynccnt(n)
#else
#define HAVE_ASYNC 0
#define WAIT_ASYNC(n)
#endif

// 16-byte global(bf16) -> LDS copy: async DMA on CDNA5, sync fallback otherwise
static __device__ __forceinline__ void cp16(const unsigned short* g, unsigned int* l) {
#if HAVE_ASYNC
  __builtin_amdgcn_global_load_async_to_lds_b128(
      (AS1 v4i_t*)(AS1 void*)(void*)g,
      (AS3 v4i_t*)(AS3 void*)l, 0, 0);
#else
  *(uint4*)l = *(const uint4*)g;
#endif
}

static __device__ __forceinline__ unsigned short f32_bf16(float f) {
  union { float f; unsigned int u; } v; v.f = f;
  unsigned int u = v.u;
  u += 0x7FFFu + ((u >> 16) & 1u);   // round-to-nearest-even
  return (unsigned short)(u >> 16);
}

static __device__ __forceinline__ v16bf frag_from(const unsigned int* base0,
                                                  const unsigned int* base1) {
  const uint4 a0 = *(const uint4*)base0;
  const uint4 a1 = *(const uint4*)base1;
  v8u u = {a0.x, a0.y, a0.z, a0.w, a1.x, a1.y, a1.z, a1.w};
  return __builtin_bit_cast(v16bf, u);
}

// ---------- tiny prep kernels ----------
__global__ void k_T(const float* __restrict__ mtx1, const float* __restrict__ mtx2,
                    float* __restrict__ T) {
  int t = blockIdx.x * blockDim.x + threadIdx.x;
  if (t >= K2_SZ * NBLK) return;
  int a = t / NBLK, l = t % NBLK;
  float s = 0.f;
  for (int m = 0; m < M_SZ; ++m)
    s += mtx1[a * M_SZ + m] * mtx2[(a * M_SZ + m) * NBLK + l];
  T[t] = s;
}

__global__ void k_S(const float* __restrict__ cparams, const float* __restrict__ conv_w,
                    const float* __restrict__ T, unsigned short* __restrict__ S) {
  int t = blockIdx.x * blockDim.x + threadIdx.x;   // 524288 total
  int ko = t >> 11, il = t & 2047;
  int k = ko >> 6, o = ko & 63;
  int i = il >> 6, l = il & 63;
  float cw[K2_SZ];
  const float* cwp = conv_w + ((size_t)o * CIN + i) * K2_SZ;
  #pragma unroll
  for (int b = 0; b < K2_SZ; ++b) cw[b] = cwp[b];
  float s = 0.f;
  for (int a = 0; a < K2_SZ; ++a) {
    const float* cp = cparams + ((size_t)k * K2_SZ + a) * K2_SZ;
    float acc = 0.f;
    #pragma unroll
    for (int b = 0; b < K2_SZ; ++b) acc += cp[b] * cw[b];
    s += acc * T[a * NBLK + l];
  }
  S[t] = f32_bf16(s);
}

__global__ void k_X(const float* __restrict__ x, unsigned short* __restrict__ Xb) {
  int t = blockIdx.x * blockDim.x + threadIdx.x;
  int base = t * 4;
  float4 v = *(const float4*)(x + base);
  Xb[base + 0] = f32_bf16(v.x);
  Xb[base + 1] = f32_bf16(v.y);
  Xb[base + 2] = f32_bf16(v.z);
  Xb[base + 3] = f32_bf16(v.w);
}

// ---------- GEMM1: R[b,q,i,l] = sum_p X[(b,i),p] * W[l,q,p] ----------
// grid (4, 448): x = M/256 ; y = l*7 + qtile ; 256 threads (8 waves)
__global__ __launch_bounds__(256) void k_gemm1(const unsigned short* __restrict__ Xb,
                                               const float* __restrict__ W,
                                               unsigned short* __restrict__ R) {
  __shared__ unsigned int As[2][256 * LDSPAD];   // 40 KB
  __shared__ unsigned int Bs[2][112 * LDSPAD];   // 17.5 KB
  const int tid  = threadIdx.x;
  const int wave = tid >> 5, lane = tid & 31;
  const int lo = lane & 15, hi = lane >> 4;
  const int m_blk = blockIdx.x * 256;
  const int l  = blockIdx.y / 7;
  const int q0 = (blockIdx.y % 7) * 112;

  v8f acc[2][7] = {};
  const unsigned short* Abase = Xb + (size_t)m_blk * P_SZ;
  const float*          Wbase = W + ((size_t)l * Q_SZ + q0) * P_SZ;

  auto stage = [&](int buf, int k0) {
    // A tile (bf16, async DMA): 256 rows x 64B = 4 x b128 per thread
    #pragma unroll
    for (int j = 0; j < 4; ++j) {
      const int c = tid + 256 * j;
      const int row = c >> 2, quad = c & 3;
      cp16(Abase + (size_t)row * P_SZ + k0 + quad * 8,
           &As[buf][row * LDSPAD + quad * 4]);
    }
    // B tile: fp32 -> bf16 on the fly (224 tasks, VALU cvt path)
    if (tid < 224) {
      const int brow = tid >> 1, bhalf = tid & 1;
      const float* src = Wbase + (size_t)brow * P_SZ + k0 + bhalf * 16;
      __builtin_prefetch(src + 8 * 32, 0, 0);   // 8 k-iterations ahead
      unsigned int d[8];
      #pragma unroll
      for (int j2 = 0; j2 < 8; ++j2) {
        float2 f = ((const float2*)src)[j2];
        d[j2] = (unsigned int)f32_bf16(f.x) | ((unsigned int)f32_bf16(f.y) << 16);
      }
      unsigned int* bdst = &Bs[buf][brow * LDSPAD + bhalf * 8];
      *(uint4*)(bdst)     = make_uint4(d[0], d[1], d[2], d[3]);
      *(uint4*)(bdst + 4) = make_uint4(d[4], d[5], d[6], d[7]);
    }
  };

  stage(0, 0);
  const int NIT = P_SZ / 32;
  for (int it = 0; it < NIT; ++it) {
    const int cur = it & 1;
    if (it + 1 < NIT) {
      stage(cur ^ 1, (it + 1) * 32);
      WAIT_ASYNC(4);          // previous tile's 4 async b128 done; next 4 in flight
    } else {
      WAIT_ASYNC(0);
    }
    __syncthreads();

    v16bf afr[2];
    #pragma unroll
    for (int s = 0; s < 2; ++s) {
      const int mrow = wave * 32 + s * 16 + lo;
      afr[s] = frag_from(&As[cur][mrow * LDSPAD + 4 * hi],
                         &As[cur][mrow * LDSPAD + 8 + 4 * hi]);
    }
    #pragma unroll
    for (int t = 0; t < 7; ++t) {
      const int n = t * 16 + lo;
      v16bf bfr = frag_from(&Bs[cur][n * LDSPAD + 4 * hi],
                            &Bs[cur][n * LDSPAD + 8 + 4 * hi]);
      acc[0][t] = __builtin_amdgcn_wmma_f32_16x16x32_bf16(false, afr[0], false, bfr,
                                                          (short)0, acc[0][t], false, false);
      acc[1][t] = __builtin_amdgcn_wmma_f32_16x16x32_bf16(false, afr[1], false, bfr,
                                                          (short)0, acc[1][t], false, false);
    }
    __syncthreads();
  }

  // scatter-store into R[b][q][i][l] (bf16)
  #pragma unroll
  for (int s = 0; s < 2; ++s) {
    const int m0 = m_blk + wave * 32 + s * 16;
    const int bb = m0 / CIN;
    const int ib = m0 % CIN;
    #pragma unroll
    for (int t = 0; t < 7; ++t) {
      const int q = q0 + t * 16 + lo;
      #pragma unroll
      for (int r = 0; r < 8; ++r) {
        const int i = ib + r + 8 * hi;
        size_t idx = (((size_t)bb * Q_SZ + q) * CIN + i) * NBLK + l;
        R[idx] = f32_bf16(acc[s][t][r]);
      }
    }
  }
}

// ---------- GEMM2: Out[(b,q),(k,o)] = R[(b,q),(i,l)] @ S[(k,o),(i,l)]^T ----------
// grid (98, 2); 256 threads (8 waves)
__global__ __launch_bounds__(256) void k_gemm2(const unsigned short* __restrict__ R,
                                               const unsigned short* __restrict__ S,
                                               const float* __restrict__ conv_b,
                                               float* __restrict__ out) {
  __shared__ unsigned int As[2][256 * LDSPAD];   // 40 KB
  __shared__ unsigned int Ss[2][128 * LDSPAD];   // 20 KB
  const int tid  = threadIdx.x;
  const int wave = tid >> 5, lane = tid & 31;
  const int lo = lane & 15, hi = lane >> 4;
  const int m_blk = blockIdx.x * 256;   // over bq
  const int n_blk = blockIdx.y * 128;   // over ko

  v8f acc[2][8] = {};
  const unsigned short* Rbase = R + (size_t)m_blk * KIL;
  const unsigned short* Sbase = S + (size_t)n_blk * KIL;

  auto stage = [&](int buf, int k0) {
    #pragma unroll
    for (int j = 0; j < 4; ++j) {      // A: 256 rows x 64B
      const int c = tid + 256 * j;
      const int row = c >> 2, quad = c & 3;
      cp16(Rbase + (size_t)row * KIL + k0 + quad * 8,
           &As[buf][row * LDSPAD + quad * 4]);
    }
    #pragma unroll
    for (int j = 0; j < 2; ++j) {      // S: 128 rows x 64B
      const int c = tid + 256 * j;
      const int row = c >> 2, quad = c & 3;
      cp16(Sbase + (size_t)row * KIL + k0 + quad * 8,
           &Ss[buf][row * LDSPAD + quad * 4]);
    }
  };

  stage(0, 0);
  const int NIT = KIL / 32;
  for (int it = 0; it < NIT; ++it) {
    const int cur = it & 1;
    if (it + 1 < NIT) {
      stage(cur ^ 1, (it + 1) * 32);
      WAIT_ASYNC(6);
    } else {
      WAIT_ASYNC(0);
    }
    __syncthreads();

    v16bf afr[2];
    #pragma unroll
    for (int s = 0; s < 2; ++s) {
      const int mrow = wave * 32 + s * 16 + lo;
      afr[s] = frag_from(&As[cur][mrow * LDSPAD + 4 * hi],
                         &As[cur][mrow * LDSPAD + 8 + 4 * hi]);
    }
    #pragma unroll
    for (int t = 0; t < 8; ++t) {
      const int n = t * 16 + lo;
      v16bf bfr = frag_from(&Ss[cur][n * LDSPAD + 4 * hi],
                            &Ss[cur][n * LDSPAD + 8 + 4 * hi]);
      acc[0][t] = __builtin_amdgcn_wmma_f32_16x16x32_bf16(false, afr[0], false, bfr,
                                                          (short)0, acc[0][t], false, false);
      acc[1][t] = __builtin_amdgcn_wmma_f32_16x16x32_bf16(false, afr[1], false, bfr,
                                                          (short)0, acc[1][t], false, false);
    }
    __syncthreads();
  }

  // epilogue: out[b][o][k][q] = acc + CIN*conv_b[o]
  #pragma unroll
  for (int t = 0; t < 8; ++t) {
    const int ko = n_blk + t * 16 + lo;
    const int k = ko >> 6, o = ko & 63;
    const float bias = (float)CIN * conv_b[o];
    #pragma unroll
    for (int s = 0; s < 2; ++s) {
      #pragma unroll
      for (int r = 0; r < 8; ++r) {
        const int m = m_blk + wave * 32 + s * 16 + r + 8 * hi;
        const int bb = m / Q_SZ, q = m % Q_SZ;
        out[(((size_t)bb * COUT + o) * 4 + k) * Q_SZ + q] = acc[s][t][r] + bias;
      }
    }
  }
}

// ---------- launcher ----------
extern "C" void kernel_launch(void* const* d_in, const int* in_sizes, int n_in,
                              void* d_out, int out_size, void* d_ws, size_t ws_size,
                              hipStream_t stream) {
  (void)in_sizes; (void)n_in; (void)out_size; (void)ws_size;
  const float* img     = (const float*)d_in[0];
  const float* weights = (const float*)d_in[1];
  const float* cparams = (const float*)d_in[2];
  const float* mtx1    = (const float*)d_in[3];
  const float* mtx2    = (const float*)d_in[4];
  const float* conv_w  = (const float*)d_in[5];
  const float* conv_b  = (const float*)d_in[6];
  float* out = (float*)d_out;

  char* ws = (char*)d_ws;
  float*          T  = (float*)ws;                                        // 6.4 KB
  unsigned short* Xb = (unsigned short*)(ws + 8192);                      // 2 MB
  unsigned short* S  = (unsigned short*)(ws + 8192 + 2u * 1024 * 1024);   // 1 MB
  unsigned short* R  = (unsigned short*)(ws + 8192 + 3u * 1024 * 1024);   // ~98 MB

  k_T    <<<dim3(7),      dim3(256), 0, stream>>>(mtx1, mtx2, T);
  k_S    <<<dim3(2048),   dim3(256), 0, stream>>>(cparams, conv_w, T, S);
  k_X    <<<dim3(1024),   dim3(256), 0, stream>>>(img, Xb);
  k_gemm1<<<dim3(4, 448), dim3(256), 0, stream>>>(Xb, weights, R);
  k_gemm2<<<dim3(98, 2),  dim3(256), 0, stream>>>(R, S, conv_b, out);
}
